// BaseMixedAttention_57028575756701
// MI455X (gfx1250) — compile-verified, matrix operand order checked by reading, not run on hardware
//
#include <hip/hip_runtime.h>
#include <hip/hip_bf16.h>

// ---------------------------------------------------------------------------
// Problem constants (from reference): B=8, T=1024, D=512, NT=77, LTXT=256,
// H=8, hd=64, TE=2048, N = NT+T = 1101.
// ---------------------------------------------------------------------------
#define BB   8
#define TT   1024
#define DD   512
#define NT   77
#define LTXT 256
#define HH   8
#define HD   64
#define TE   2048
#define NKEY 1101

typedef __attribute__((ext_vector_type(16))) __bf16 v16bf;
typedef __attribute__((ext_vector_type(8)))  float  v8f;
typedef __attribute__((ext_vector_type(8)))  __bf16 bf16x8;   // 16-byte staging vector

// ---- gfx1250 async global->LDS path (ASYNCcnt), with safe fallback --------
#if defined(__gfx1250__) && \
    __has_builtin(__builtin_amdgcn_global_load_async_to_lds_b128) && \
    __has_builtin(__builtin_amdgcn_s_wait_asynccnt)
#define HAVE_ASYNC_LDS 1
typedef __attribute__((ext_vector_type(4))) int i4v;          // 16-byte payload
typedef __attribute__((address_space(1))) i4v gi4v;           // global int4
typedef __attribute__((address_space(3))) i4v li4v;           // LDS int4
#define ASYNC_LDS_B128(gp, lp) \
    __builtin_amdgcn_global_load_async_to_lds_b128((gi4v*)(gp), (li4v*)(lp), 0, 0)
#define WAIT_ASYNC() __builtin_amdgcn_s_wait_asynccnt(0)
#else
#define HAVE_ASYNC_LDS 0
#define WAIT_ASYNC() ((void)0)
#endif

// K-index pattern for 16-bit WMMA A/B fragments (16x32 / 32x16), per ISA
// 7.12.2: lanes 0-15 hold K = {0..7, 16..23}; lanes 16-31 hold K = {8..15, 24..31}.
__device__ __forceinline__ int kpat(int e, int lane) {
    int base = (lane < 16) ? 0 : 8;
    return (e < 8) ? (base + e) : (base + e + 8);
}

__device__ __forceinline__ float silu_f(float x) {
    return x / (1.0f + __expf(-x));
}

// ---------------------------------------------------------------------------
// LayerNorm of fp32 rows -> bf16 output rows (used for xn and xfn).
// ---------------------------------------------------------------------------
__global__ __launch_bounds__(256)
void ln_to_bf16(const float* __restrict__ x, const float* __restrict__ g,
                const float* __restrict__ bta, __bf16* __restrict__ out, int Dw) {
    const int row = blockIdx.x;
    const int tid = threadIdx.x;
    const float* xr = x + (size_t)row * Dw;
    __shared__ float red0[8], red1[8];

    float s = 0.f, s2 = 0.f;
    for (int i = tid; i < Dw; i += 256) { float v = xr[i]; s += v; s2 += v * v; }
    for (int off = 16; off > 0; off >>= 1) {
        s  += __shfl_xor(s,  off, 32);
        s2 += __shfl_xor(s2, off, 32);
    }
    if ((tid & 31) == 0) { red0[tid >> 5] = s; red1[tid >> 5] = s2; }
    __syncthreads();
    if (tid == 0) {
        float a = 0.f, b2 = 0.f;
        for (int i = 0; i < 8; i++) { a += red0[i]; b2 += red1[i]; }
        red0[0] = a; red1[0] = b2;
    }
    __syncthreads();
    const float mu  = red0[0] / (float)Dw;
    const float var = red1[0] / (float)Dw - mu * mu;
    const float rs  = rsqrtf(var + 1e-5f);
    for (int i = tid; i < Dw; i += 256)
        out[(size_t)row * Dw + i] = (__bf16)((xr[i] - mu) * rs * g[i] + bta[i]);
}

// ---------------------------------------------------------------------------
// emb_out[b, j] = sum_k silu(emb[b,k]) * emb_W[j,k] + emb_b[j]   (B x 1024)
// ---------------------------------------------------------------------------
__global__ __launch_bounds__(256)
void emb_proj(const float* __restrict__ emb, const float* __restrict__ W,
              const float* __restrict__ bias, float* __restrict__ eo) {
    const int j = blockIdx.x * 256 + threadIdx.x;   // 0 .. 8191
    const int b = j >> 10;
    const int c = j & 1023;
    const float* er = emb + (size_t)b * TE;
    const float* wr = W + (size_t)c * TE;
    float acc = bias[c];
    for (int k = 0; k < TE; k++) acc += silu_f(er[k]) * wr[k];
    eo[j] = acc;
}

// ---------------------------------------------------------------------------
// WMMA GEMM: C[m,n] = A[m,:K] . W[n,:K] + bias[n], epilogue variants.
//   mode 0 : write bf16
//   mode 1 : write bf16, scale by text_cond(cond_type[batch])
//   mode 2 : write bf16, scale by rowscale[m] (src_mask)
//   mode 3 : write fp32, add resid[m*512+n] (residual output)
// Block: 256 thr = 8 waves; tile 128x128x32; wave tile 64x32 (4x2 WMMA accum).
// A-tile is staged via async global->LDS DMA (ASYNCcnt) when available.
// ---------------------------------------------------------------------------
__global__ __launch_bounds__(256)
void gemm_wmma(const __bf16* __restrict__ A, int lda,
               const float* __restrict__ W, const float* __restrict__ bias,
               void* __restrict__ Cout,
               const float* __restrict__ resid,
               const float* __restrict__ rowscale,
               const int* __restrict__ cond_type,
               int Mtot, int K, int rows_per_batch, int out_row_off,
               long long out_batch_stride, int mode) {
    const int m0   = blockIdx.x * 128;
    const int n0   = blockIdx.y * 128;
    const int tid  = threadIdx.x;
    const int lane = tid & 31;
    const int w    = tid >> 5;

    __shared__ __attribute__((aligned(16))) __bf16 As[128][40];  // [m][k], 80B pitch
    __shared__ __attribute__((aligned(16))) __bf16 Bs[128][40];  // [n][k]

    const int mrow = (w & 1) * 64;   // 2 wave-rows  x 64
    const int ncol = (w >> 1) * 32;  // 4 wave-cols  x 32

    v8f acc[4][2];
    {
        v8f z = {};
        for (int mi = 0; mi < 4; mi++) { acc[mi][0] = z; acc[mi][1] = z; }
    }

    const int r  = tid >> 1;          // 0..127
    const int kb = (tid & 1) * 16;    // 0 or 16

    for (int k0 = 0; k0 < K; k0 += 32) {
        __syncthreads();
        // ---- stage A tile (bf16): async DMA to LDS when available ----
        {
            const int gm = m0 + r;
            const __bf16* ap = A + (size_t)gm * lda + k0 + kb;
#if HAVE_ASYNC_LDS
            if (gm < Mtot) {
                ASYNC_LDS_B128(ap,     &As[r][kb]);
                ASYNC_LDS_B128(ap + 8, &As[r][kb + 8]);
                if (k0 + 32 < K) __builtin_prefetch(ap + 32, 0, 3);
            } else {
                bf16x8 z = {};
                *(bf16x8*)&As[r][kb]     = z;
                *(bf16x8*)&As[r][kb + 8] = z;
            }
#else
            bf16x8 a0 = {}, a1 = {};
            if (gm < Mtot) {
                a0 = ((const bf16x8*)ap)[0];
                a1 = ((const bf16x8*)ap)[1];
                if (k0 + 32 < K) __builtin_prefetch(ap + 32, 0, 3);
            }
            *(bf16x8*)&As[r][kb]     = a0;
            *(bf16x8*)&As[r][kb + 8] = a1;
#endif
        }
        // ---- stage W tile: float4 x4 loads, cvt to bf16, 2x ds_store_b128 ----
        {
            const float* wr = W + (size_t)(n0 + r) * K + k0 + kb;
            const float4* wp = (const float4*)wr;
            float4 f0 = wp[0], f1 = wp[1], f2 = wp[2], f3 = wp[3];
            bf16x8 w0, w1;
            w0[0] = (__bf16)f0.x; w0[1] = (__bf16)f0.y;
            w0[2] = (__bf16)f0.z; w0[3] = (__bf16)f0.w;
            w0[4] = (__bf16)f1.x; w0[5] = (__bf16)f1.y;
            w0[6] = (__bf16)f1.z; w0[7] = (__bf16)f1.w;
            w1[0] = (__bf16)f2.x; w1[1] = (__bf16)f2.y;
            w1[2] = (__bf16)f2.z; w1[3] = (__bf16)f2.w;
            w1[4] = (__bf16)f3.x; w1[5] = (__bf16)f3.y;
            w1[6] = (__bf16)f3.z; w1[7] = (__bf16)f3.w;
            *(bf16x8*)&Bs[r][kb]     = w0;
            *(bf16x8*)&Bs[r][kb + 8] = w1;
            if (k0 + 32 < K) __builtin_prefetch(wr + 32, 0, 3);
        }
        WAIT_ASYNC();
        __syncthreads();

        // ---- WMMA: one K-step of 32 per staged tile ----
        v16bf bfrag[2];
        {
            const int nb = ncol + (lane & 15);
            for (int e = 0; e < 16; e++) {
                const int kd = kpat(e, lane);
                bfrag[0][e] = Bs[nb][kd];
                bfrag[1][e] = Bs[nb + 16][kd];
            }
        }
        for (int mi = 0; mi < 4; mi++) {
            v16bf a;
            const int mm = mrow + mi * 16 + (lane & 15);
            for (int e = 0; e < 16; e++) a[e] = As[mm][kpat(e, lane)];
            acc[mi][0] = __builtin_amdgcn_wmma_f32_16x16x32_bf16(
                false, a, false, bfrag[0], (short)0, acc[mi][0], false, false);
            acc[mi][1] = __builtin_amdgcn_wmma_f32_16x16x32_bf16(
                false, a, false, bfrag[1], (short)0, acc[mi][1], false, false);
        }
    }

    // ---- epilogue ----
    const int rofs = (lane < 16) ? 0 : 8;
    const int c    = lane & 15;
    for (int mi = 0; mi < 4; mi++) {
        for (int nj = 0; nj < 2; nj++) {
            for (int jv = 0; jv < 8; jv++) {
                const int m = m0 + mrow + mi * 16 + jv + rofs;
                if (m >= Mtot) continue;
                const int n = n0 + ncol + nj * 16 + c;
                float val = acc[mi][nj][jv] + bias[n];
                const int bb  = m / rows_per_batch;
                const int loc = m % rows_per_batch;
                const size_t oidx = (size_t)bb * (size_t)out_batch_stride +
                                    (size_t)(out_row_off + loc) * DD + n;
                if (mode == 1) {
                    val *= ((cond_type[bb] % 10) > 0) ? 1.0f : 0.0f;
                } else if (mode == 2) {
                    val *= rowscale[m];
                }
                if (mode == 3) {
                    ((float*)Cout)[oidx] = val + resid[(size_t)m * DD + n];
                } else {
                    ((__bf16*)Cout)[oidx] = (__bf16)val;
                }
            }
        }
    }
}

// ---------------------------------------------------------------------------
// Flash-attention (no 1/sqrt(d) scale, matching reference) per (b, h, 64-q tile).
// Streams 64-key chunks of K/V through LDS (async DMA when available);
// online softmax; QK^T and PV via WMMA. 256 threads = 8 waves.
// ---------------------------------------------------------------------------
__global__ __launch_bounds__(256)
void attn_wmma(const __bf16* __restrict__ q, const __bf16* __restrict__ k,
               const __bf16* __restrict__ v, const float* __restrict__ src_mask,
               const int* __restrict__ cond_type, float* __restrict__ y) {
    const int t0 = blockIdx.x * 64;
    const int h  = blockIdx.y;
    const int b  = blockIdx.z;
    const int tid  = threadIdx.x;
    const int lane = tid & 31;
    const int w    = tid >> 5;

    __shared__ __attribute__((aligned(16))) __bf16 Qs[64][72];  // 144B pitch
    __shared__ __attribute__((aligned(16))) __bf16 Ks[64][72];
    __shared__ __attribute__((aligned(16))) __bf16 Vs[64][72];
    __shared__ __attribute__((aligned(16))) __bf16 Ps[64][72];
    __shared__ float  Ss[64][64];
    __shared__ float  rowm[64], rowl[64], alpha[64];

    // Load Q tile: 64x64 elems = 512 x bf16x8, 2 per thread.
    for (int i = tid; i < 512; i += 256) {
        const int rq = i >> 3;
        const int d8 = (i & 7) * 8;
        const __bf16* qp = q + ((size_t)(b * TT + t0 + rq) * DD) + h * HD + d8;
#if HAVE_ASYNC_LDS
        ASYNC_LDS_B128(qp, &Qs[rq][d8]);
#else
        *(bf16x8*)&Qs[rq][d8] = *(const bf16x8*)qp;
#endif
    }
    if (tid < 64) { rowm[tid] = -1e30f; rowl[tid] = 0.f; }

    const float tc = ((cond_type[b] % 10) > 0) ? 1.0f : 0.0f;

    const int qr = (w & 3) * 16;
    const int wc = (w >> 2) * 32;

    v8f o0 = {}, o1 = {};

    for (int n0 = 0; n0 < NKEY; n0 += 64) {
        __syncthreads();
        // ---- stage K/V chunk ----
        for (int i = tid; i < 512; i += 256) {
            const int rr = i >> 3;
            const int d8 = (i & 7) * 8;
            const int key = n0 + rr;
#if HAVE_ASYNC_LDS
            if (key < NKEY) {
                const size_t idx = ((size_t)(b * NKEY + key) * DD) + h * HD + d8;
                ASYNC_LDS_B128(k + idx, &Ks[rr][d8]);
                ASYNC_LDS_B128(v + idx, &Vs[rr][d8]);
                if (key + 64 < NKEY)
                    __builtin_prefetch(k + idx + (size_t)64 * DD, 0, 3);
            } else {
                bf16x8 z = {};
                *(bf16x8*)&Ks[rr][d8] = z;
                *(bf16x8*)&Vs[rr][d8] = z;
            }
#else
            bf16x8 kv = {}, vv = {};
            if (key < NKEY) {
                const size_t idx = ((size_t)(b * NKEY + key) * DD) + h * HD + d8;
                kv = *(const bf16x8*)(k + idx);
                vv = *(const bf16x8*)(v + idx);
                if (key + 64 < NKEY)
                    __builtin_prefetch(k + idx + (size_t)64 * DD, 0, 3);
            }
            *(bf16x8*)&Ks[rr][d8] = kv;
            *(bf16x8*)&Vs[rr][d8] = vv;
#endif
        }
        WAIT_ASYNC();                 // wave's async DMAs done (covers Q on 1st iter)
        __syncthreads();

        // ---- S = Q . K^T (this wave: 16 q-rows x 32 keys) ----
        v8f s0 = {}, s1 = {};
        for (int kk = 0; kk < HD; kk += 32) {
            v16bf a, b0f, b1f;
            const int mq = qr + (lane & 15);
            const int nk = wc + (lane & 15);
            for (int e = 0; e < 16; e++) {
                const int kd = kpat(e, lane) + kk;
                a[e]   = Qs[mq][kd];
                b0f[e] = Ks[nk][kd];        // B[k=d][n=key]
                b1f[e] = Ks[nk + 16][kd];
            }
            s0 = __builtin_amdgcn_wmma_f32_16x16x32_bf16(false, a, false, b0f,
                                                         (short)0, s0, false, false);
            s1 = __builtin_amdgcn_wmma_f32_16x16x32_bf16(false, a, false, b1f,
                                                         (short)0, s1, false, false);
        }
        {
            const int rofs = (lane < 16) ? 0 : 8;
            const int c = lane & 15;
            for (int jv = 0; jv < 8; jv++) {
                Ss[qr + jv + rofs][wc + c]      = s0[jv];
                Ss[qr + jv + rofs][wc + 16 + c] = s1[jv];
            }
        }
        __syncthreads();

        // ---- online softmax bookkeeping (one thread per q-row) ----
        if (tid < 64) {
            const int rr = tid;
            float cmax = -1e30f;
            for (int j = 0; j < 64; j++) {
                const int key = n0 + j;
                float s;
                if (key >= NKEY) {
                    s = -1e30f;                       // padded key -> exp() == 0
                } else {
                    const float mval = (key < NT) ? tc : src_mask[b * TT + (key - NT)];
                    s = Ss[rr][j] + (1.0f - mval) * -1000000.0f;
                }
                Ss[rr][j] = s;
                cmax = fmaxf(cmax, s);
            }
            const float nm = fmaxf(rowm[rr], cmax);
            const float al = __expf(rowm[rr] - nm);
            float sum = 0.f;
            for (int j = 0; j < 64; j++) {
                const float p = __expf(Ss[rr][j] - nm);
                Ps[rr][j] = (__bf16)p;
                sum += p;
            }
            rowl[rr]  = rowl[rr] * al + sum;
            rowm[rr]  = nm;
            alpha[rr] = al;
        }
        __syncthreads();

        // ---- rescale O and accumulate P . V (16 q-rows x 32 dims per wave) ----
        {
            const int rofs = (lane < 16) ? 0 : 8;
            for (int jv = 0; jv < 8; jv++) {
                const float al = alpha[qr + jv + rofs];
                o0[jv] *= al;
                o1[jv] *= al;
            }
        }
        for (int kk = 0; kk < 64; kk += 32) {
            v16bf a, b0f, b1f;
            const int mq = qr + (lane & 15);
            const int nd = wc + (lane & 15);
            for (int e = 0; e < 16; e++) {
                const int kd = kpat(e, lane) + kk;
                a[e]   = Ps[mq][kd];
                b0f[e] = Vs[kd][nd];          // B[k=key][n=d]
                b1f[e] = Vs[kd][nd + 16];
            }
            o0 = __builtin_amdgcn_wmma_f32_16x16x32_bf16(false, a, false, b0f,
                                                         (short)0, o0, false, false);
            o1 = __builtin_amdgcn_wmma_f32_16x16x32_bf16(false, a, false, b1f,
                                                         (short)0, o1, false, false);
        }
    }

    __syncthreads();
    // ---- normalize and write y (fp32 [B,T,D]) ----
    {
        const int rofs = (lane < 16) ? 0 : 8;
        const int c = lane & 15;
        for (int jv = 0; jv < 8; jv++) {
            const int rr = qr + jv + rofs;
            const float inv = 1.0f / rowl[rr];
            const size_t base = ((size_t)(b * TT + t0 + rr) * DD) + h * HD;
            y[base + wc + c]      = o0[jv] * inv;
            y[base + wc + 16 + c] = o1[jv] * inv;
        }
    }
}

// ---------------------------------------------------------------------------
// Stylization pre-GEMM: hln = silu( ln(y, snorm_g, snorm_b)*(1+scale) + shift )
// ---------------------------------------------------------------------------
__global__ __launch_bounds__(256)
void stylize(const float* __restrict__ y, const float* __restrict__ eo,
             const float* __restrict__ sg, const float* __restrict__ sb,
             __bf16* __restrict__ out) {
    const int row = blockIdx.x;          // 0..B*T-1
    const int b   = row >> 10;
    const int tid = threadIdx.x;
    const float* yr = y + (size_t)row * DD;
    __shared__ float red0[8], red1[8];

    float s = 0.f, s2 = 0.f;
    for (int i = tid; i < DD; i += 256) { float vv = yr[i]; s += vv; s2 += vv * vv; }
    for (int off = 16; off > 0; off >>= 1) {
        s  += __shfl_xor(s,  off, 32);
        s2 += __shfl_xor(s2, off, 32);
    }
    if ((tid & 31) == 0) { red0[tid >> 5] = s; red1[tid >> 5] = s2; }
    __syncthreads();
    if (tid == 0) {
        float a = 0.f, b2 = 0.f;
        for (int i = 0; i < 8; i++) { a += red0[i]; b2 += red1[i]; }
        red0[0] = a; red1[0] = b2;
    }
    __syncthreads();
    const float mu  = red0[0] / (float)DD;
    const float var = red1[0] / (float)DD - mu * mu;
    const float rs  = rsqrtf(var + 1e-5f);
    for (int i = tid; i < DD; i += 256) {
        const float scale = eo[b * 1024 + i];
        const float shift = eo[b * 1024 + DD + i];
        float hv = (yr[i] - mu) * rs * sg[i] + sb[i];
        hv = hv * (1.0f + scale) + shift;
        out[(size_t)row * DD + i] = (__bf16)silu_f(hv);
    }
}

// ---------------------------------------------------------------------------
// Host-side orchestration
// ---------------------------------------------------------------------------
extern "C" void kernel_launch(void* const* d_in, const int* in_sizes, int n_in,
                              void* d_out, int out_size, void* d_ws, size_t ws_size,
                              hipStream_t stream) {
    (void)in_sizes; (void)n_in; (void)out_size; (void)ws_size;

    const float* x        = (const float*)d_in[0];
    const float* xf       = (const float*)d_in[1];
    const float* emb      = (const float*)d_in[2];
    const float* src_mask = (const float*)d_in[3];
    const int*   cond     = (const int*)  d_in[4];
    const float* norm_g   = (const float*)d_in[5];
    const float* norm_b   = (const float*)d_in[6];
    const float* tnorm_g  = (const float*)d_in[7];
    const float* tnorm_b  = (const float*)d_in[8];
    const float* Wq   = (const float*)d_in[9];
    const float* bq   = (const float*)d_in[10];
    const float* Wkt  = (const float*)d_in[11];
    const float* bkt  = (const float*)d_in[12];
    const float* Wvt  = (const float*)d_in[13];
    const float* bvt  = (const float*)d_in[14];
    const float* Wkm  = (const float*)d_in[15];
    const float* bkm  = (const float*)d_in[16];
    const float* Wvm  = (const float*)d_in[17];
    const float* bvm  = (const float*)d_in[18];
    const float* embW = (const float*)d_in[19];
    const float* embB = (const float*)d_in[20];
    const float* sng  = (const float*)d_in[21];
    const float* snb  = (const float*)d_in[22];
    const float* outW = (const float*)d_in[23];
    const float* outB = (const float*)d_in[24];
    float* out = (float*)d_out;

    // Workspace layout (bytes)
    char* ws = (char*)d_ws;
    const size_t off_eo  = 0;                                   // B*1024 f32
    const size_t off_xn  = off_eo  + (size_t)BB * 1024 * 4;     // B*T*D bf16
    const size_t off_xfn = off_xn  + (size_t)BB * TT * DD * 2;  // B*NT*L bf16
    const size_t off_q   = off_xfn + (size_t)BB * NT * LTXT * 2;
    const size_t off_k   = off_q   + (size_t)BB * TT * DD * 2;
    const size_t off_v   = off_k   + (size_t)BB * NKEY * DD * 2;
    const size_t off_y   = off_v   + (size_t)BB * NKEY * DD * 2;
    const size_t off_hln = off_xn;                              // reuse xn (dead after GEMMs)

    float*  eo  = (float*) (ws + off_eo);
    __bf16* xn  = (__bf16*)(ws + off_xn);
    __bf16* xfn = (__bf16*)(ws + off_xfn);
    __bf16* qb  = (__bf16*)(ws + off_q);
    __bf16* kb  = (__bf16*)(ws + off_k);
    __bf16* vb  = (__bf16*)(ws + off_v);
    float*  yb  = (float*) (ws + off_y);
    __bf16* hln = (__bf16*)(ws + off_hln);

    const long long stride_qy = (long long)TT * DD;      // [B,T,D] mapping
    const long long stride_kv = (long long)NKEY * DD;    // [B,N,D] mapping

    // 1) embedding MLP (independent)
    emb_proj<<<dim3((BB * 1024) / 256), dim3(256), 0, stream>>>(emb, embW, embB, eo);

    // 2) LayerNorms -> bf16
    ln_to_bf16<<<dim3(BB * TT), dim3(256), 0, stream>>>(x,  norm_g,  norm_b,  xn,  DD);
    ln_to_bf16<<<dim3(BB * NT), dim3(256), 0, stream>>>(xf, tnorm_g, tnorm_b, xfn, LTXT);

    // 3) Projections (WMMA GEMMs), all produce bf16
    const dim3 gBig((BB * TT + 127) / 128, DD / 128);    // 64 x 4
    const dim3 gTxt((BB * NT + 127) / 128, DD / 128);    // 5 x 4
    // q = xn @ Wq^T + bq            -> [B,T,D]
    gemm_wmma<<<gBig, 256, 0, stream>>>(xn, DD, Wq, bq, qb, nullptr, nullptr, nullptr,
                                        BB * TT, DD, TT, 0, stride_qy, 0);
    // k_text = xfn @ Wkt^T + bkt    -> k[:, 0:77, :]
    gemm_wmma<<<gTxt, 256, 0, stream>>>(xfn, LTXT, Wkt, bkt, kb, nullptr, nullptr, nullptr,
                                        BB * NT, LTXT, NT, 0, stride_kv, 0);
    // k_motion = xn @ Wkm^T + bkm   -> k[:, 77:, :]
    gemm_wmma<<<gBig, 256, 0, stream>>>(xn, DD, Wkm, bkm, kb, nullptr, nullptr, nullptr,
                                        BB * TT, DD, TT, NT, stride_kv, 0);
    // v_text = (xfn @ Wvt^T + bvt) * text_cond   -> v[:, 0:77, :]
    gemm_wmma<<<gTxt, 256, 0, stream>>>(xfn, LTXT, Wvt, bvt, vb, nullptr, nullptr, cond,
                                        BB * NT, LTXT, NT, 0, stride_kv, 1);
    // v_motion = (xn @ Wvm^T + bvm) * src_mask   -> v[:, 77:, :]
    gemm_wmma<<<gBig, 256, 0, stream>>>(xn, DD, Wvm, bvm, vb, nullptr, src_mask, nullptr,
                                        BB * TT, DD, TT, NT, stride_kv, 2);

    // 4) Attention -> y (fp32)
    attn_wmma<<<dim3(TT / 64, HH, BB), 256, 0, stream>>>(qb, kb, vb, src_mask, cond, yb);

    // 5) Stylization: hln = silu(ln(y)*(1+scale)+shift)  (bf16)
    stylize<<<dim3(BB * TT), dim3(256), 0, stream>>>(yb, eo, sng, snb, hln);

    // 6) out = x + hln @ out_W^T + out_b  (fp32, residual epilogue)
    gemm_wmma<<<gBig, 256, 0, stream>>>(hln, DD, outW, outB, out, x, nullptr, nullptr,
                                        BB * TT, DD, TT, 0, stride_qy, 3);
}
